// AttentionModule_19335942766946
// MI455X (gfx1250) — matrix operand order repaired and checked by reading, not compiled, not run on hardware
//
#include <hip/hip_runtime.h>

// ---------------------------------------------------------------------------
// Problem constants (from reference): B=2, T=2048, D=1024, N=2, IS=512
// ---------------------------------------------------------------------------
#define T_DIM 2048
#define D_DIM 1024
#define IS_DIM 512
#define BNK 4  // B * N_concepts

typedef __attribute__((ext_vector_type(16))) __bf16 v16bf;
typedef __attribute__((ext_vector_type(8)))  __bf16 v8bf;
typedef __attribute__((ext_vector_type(4)))  __bf16 v4bf;
typedef __attribute__((ext_vector_type(8)))  float  v8f;

// round-to-nearest-even f32 -> bf16
static __device__ __forceinline__ __bf16 f2bf(float f) {
  unsigned int u = __float_as_uint(f);
  u += 0x7FFFu + ((u >> 16) & 1u);
  unsigned short s = (unsigned short)(u >> 16);
  __bf16 r;
  __builtin_memcpy(&r, &s, 2);
  return r;
}

// 16x32 bf16 A-fragment for v_wmma_f32_16x16x32_bf16 from a row-major matrix.
// Lane L holds row M=L%16; 16 contiguous K starting at base+(hi?16:0).
static __device__ __forceinline__ v16bf load_a_frag(const __bf16* rowptr, int hi) {
  const __bf16* p = rowptr + hi * 16;
  v8bf lo = *reinterpret_cast<const v8bf*>(p);
  v8bf hh = *reinterpret_cast<const v8bf*>(p + 8);
  v16bf r;
#pragma unroll
  for (int j = 0; j < 8; ++j) { r[j] = lo[j]; r[j + 8] = hh[j]; }
  return r;
}

// B-fragment (32x16): lane L holds column N=L%16 with 16 consecutive K values
// starting at (hi?16:0) -> 16 contiguous halves from a K-contiguous row.
static __device__ __forceinline__ v16bf load_b_frag_contig(const __bf16* p) {
  v8bf lo = *reinterpret_cast<const v8bf*>(p);
  v8bf hh = *reinterpret_cast<const v8bf*>(p + 8);
  v16bf r;
#pragma unroll
  for (int j = 0; j < 8; ++j) { r[j] = lo[j]; r[j + 8] = hh[j]; }
  return r;
}

// ---------------------------------------------------------------------------
// Kernel 0: f32 -> bf16 elementwise conversion (row-major copy of x)
// ---------------------------------------------------------------------------
__global__ __launch_bounds__(256) void cvt_bf16(const float* __restrict__ src,
                                                __bf16* __restrict__ dst, int n) {
  int i = blockIdx.x * 256 + threadIdx.x;
  if (i < n) dst[i] = f2bf(src[i]);
}

// ---------------------------------------------------------------------------
// Kernel 1: batched 32x32-tiled transpose + f32->bf16:  dst[c][r] = src[r][c]
// Tile staged memory->LDS with gfx1250 async-to-LDS copies (ASYNCcnt),
// then read transposed out of padded LDS.
// Grid: (C/32, R/32, batch); 256 threads (each moves 16B in, 8B out).
// ---------------------------------------------------------------------------
__global__ __launch_bounds__(256) void transpose_cvt(const float* __restrict__ src,
                                                     __bf16* __restrict__ dst,
                                                     int R, int C) {
  __shared__ float tile[32 * 36];  // stride 36 floats (bank-conflict pad)
  const size_t bstride = (size_t)R * C;
  src += (size_t)blockIdx.z * bstride;
  dst += (size_t)blockIdx.z * bstride;
  const int c0 = blockIdx.x * 32;
  const int r0 = blockIdx.y * 32;
  const int tid = threadIdx.x;
  const int rr = tid >> 3;         // 0..31  (source row within tile)
  const int cc = (tid & 7) * 4;    // 0,4,...,28 (source col base, 4 floats)

  {  // async global -> LDS copy of the 32x32 f32 tile (row-major, stride 36)
    const unsigned lds_addr =
        (unsigned)(size_t)(void*)&tile[rr * 36 + cc];  // addr[31:0] = LDS offset
    const unsigned long long gaddr =
        (unsigned long long)(size_t)(const void*)&src[(size_t)(r0 + rr) * C + c0 + cc];
    asm volatile("global_load_async_to_lds_b128 %0, %1, off"
                 :: "v"(lds_addr), "v"(gaddr)
                 : "memory");
  }
  asm volatile("s_wait_asynccnt 0x0" ::: "memory");
  __syncthreads();

  // transposed read: output row d = c0+rr, output cols r0+cc..+4
  v4bf o;
#pragma unroll
  for (int j = 0; j < 4; ++j) o[j] = f2bf(tile[(cc + j) * 36 + rr]);
  *reinterpret_cast<v4bf*>(&dst[(size_t)(c0 + rr) * R + r0 + cc]) = o;
}

// ---------------------------------------------------------------------------
// Kernel 2: WQ[bn] = X[b] @ W[n]  (2048x1024 @ 1024x512, bf16 out)
// A = xbf row-major direct; B = WT (w transposed, [h][d]) direct.
// Block tile 128(M) x 64(N), 8 waves (4x2), wave tile 32x32, K step 32.
// ---------------------------------------------------------------------------
__global__ __launch_bounds__(256) void gemm_wq(const __bf16* __restrict__ xbf,
                                               const __bf16* __restrict__ wt,
                                               __bf16* __restrict__ wqbf) {
  const int bn = blockIdx.z;
  const int b = bn >> 1, n = bn & 1;
  const int m0 = blockIdx.x * 128;
  const int n0 = blockIdx.y * 64;
  const __bf16* X = xbf + (size_t)b * T_DIM * D_DIM;
  const __bf16* WT = wt + (size_t)n * IS_DIM * D_DIM;  // [h][d]
  __bf16* O = wqbf + (size_t)bn * T_DIM * IS_DIM;

  const int lane = threadIdx.x & 31;
  const int wave = threadIdx.x >> 5;
  const int wr = wave >> 1, wc = wave & 1;
  const int lm = lane & 15, hi = lane >> 4;

  v8f acc[2][2];
#pragma unroll
  for (int i = 0; i < 2; ++i)
#pragma unroll
    for (int j = 0; j < 2; ++j) acc[i][j] = v8f{0.f, 0.f, 0.f, 0.f, 0.f, 0.f, 0.f, 0.f};

  for (int k0 = 0; k0 < D_DIM; k0 += 32) {
    v16bf afr[2], bfr[2];
#pragma unroll
    for (int mi = 0; mi < 2; ++mi) {
      const int row = m0 + wr * 32 + mi * 16 + lm;
      afr[mi] = load_a_frag(&X[(size_t)row * D_DIM + k0], hi);
    }
#pragma unroll
    for (int ni = 0; ni < 2; ++ni) {
      const int h = n0 + wc * 32 + ni * 16 + lm;
      bfr[ni] = load_b_frag_contig(&WT[(size_t)h * D_DIM + k0 + hi * 16]);
    }
#pragma unroll
    for (int mi = 0; mi < 2; ++mi)
#pragma unroll
      for (int ni = 0; ni < 2; ++ni)
        acc[mi][ni] = __builtin_amdgcn_wmma_f32_16x16x32_bf16(
            false, afr[mi], false, bfr[ni], (short)0, acc[mi][ni], false, false);
  }

#pragma unroll
  for (int mi = 0; mi < 2; ++mi)
#pragma unroll
    for (int ni = 0; ni < 2; ++ni)
#pragma unroll
      for (int r = 0; r < 8; ++r) {
        const int row = m0 + wr * 32 + mi * 16 + (hi ? r + 8 : r);
        const int col = n0 + wc * 32 + ni * 16 + lm;
        O[(size_t)row * IS_DIM + col] = f2bf(acc[mi][ni][r]);
      }
}

// ---------------------------------------------------------------------------
// Kernel 3: S[bn] = WQ[bn] @ WQ[bn]^T  (f32 logits -> attn region of d_out)
// B = WQ^T: B-fragment column N=t with contiguous K is row t of row-major WQ.
// ---------------------------------------------------------------------------
__global__ __launch_bounds__(256) void gemm_attn_logits(const __bf16* __restrict__ wqbf,
                                                        float* __restrict__ attn_out) {
  const int bn = blockIdx.z;
  const int s0 = blockIdx.x * 128;
  const int t0 = blockIdx.y * 64;
  const __bf16* Wq = wqbf + (size_t)bn * T_DIM * IS_DIM;
  float* S = attn_out + (size_t)bn * T_DIM * T_DIM;

  const int lane = threadIdx.x & 31;
  const int wave = threadIdx.x >> 5;
  const int wr = wave >> 1, wc = wave & 1;
  const int lm = lane & 15, hi = lane >> 4;

  v8f acc[2][2];
#pragma unroll
  for (int i = 0; i < 2; ++i)
#pragma unroll
    for (int j = 0; j < 2; ++j) acc[i][j] = v8f{0.f, 0.f, 0.f, 0.f, 0.f, 0.f, 0.f, 0.f};

  for (int k0 = 0; k0 < IS_DIM; k0 += 32) {
    if (k0 + 32 < IS_DIM) {  // global_prefetch_b8 of next K slab
      __builtin_prefetch(&Wq[(size_t)(s0 + wr * 32 + lm) * IS_DIM + k0 + 32], 0, 1);
      __builtin_prefetch(&Wq[(size_t)(t0 + wc * 32 + lm) * IS_DIM + k0 + 32], 0, 1);
    }
    v16bf afr[2], bfr[2];
#pragma unroll
    for (int mi = 0; mi < 2; ++mi) {
      const int row = s0 + wr * 32 + mi * 16 + lm;
      afr[mi] = load_a_frag(&Wq[(size_t)row * IS_DIM + k0], hi);
    }
#pragma unroll
    for (int ni = 0; ni < 2; ++ni) {
      const int trow = t0 + wc * 32 + ni * 16 + lm;
      bfr[ni] = load_b_frag_contig(&Wq[(size_t)trow * IS_DIM + k0 + hi * 16]);
    }
#pragma unroll
    for (int mi = 0; mi < 2; ++mi)
#pragma unroll
      for (int ni = 0; ni < 2; ++ni)
        acc[mi][ni] = __builtin_amdgcn_wmma_f32_16x16x32_bf16(
            false, afr[mi], false, bfr[ni], (short)0, acc[mi][ni], false, false);
  }

#pragma unroll
  for (int mi = 0; mi < 2; ++mi)
#pragma unroll
    for (int ni = 0; ni < 2; ++ni)
#pragma unroll
      for (int r = 0; r < 8; ++r) {
        const int row = s0 + wr * 32 + mi * 16 + (hi ? r + 8 : r);
        const int col = t0 + wc * 32 + ni * 16 + lm;
        S[(size_t)row * T_DIM + col] = acc[mi][ni][r];
      }
}

// ---------------------------------------------------------------------------
// Kernel 4: row softmax in place on attn (f32) + bf16 copy of probs to ws.
// One block per 2048-wide row; each thread owns 8 columns.
// ---------------------------------------------------------------------------
__global__ __launch_bounds__(256) void softmax_rows(float* __restrict__ attn,
                                                    __bf16* __restrict__ pbf) {
  __shared__ float red[256];
  const int row = blockIdx.x;
  float* p = attn + (size_t)row * T_DIM;
  __bf16* q = pbf + (size_t)row * T_DIM;
  const int tid = threadIdx.x;

  float v[8];
  float mx = -3.402823e38f;
#pragma unroll
  for (int i = 0; i < 8; ++i) { v[i] = p[tid + i * 256]; mx = fmaxf(mx, v[i]); }
  red[tid] = mx;
  __syncthreads();
  for (int s = 128; s > 0; s >>= 1) {
    if (tid < s) red[tid] = fmaxf(red[tid], red[tid + s]);
    __syncthreads();
  }
  mx = red[0];
  __syncthreads();

  float sum = 0.f;
#pragma unroll
  for (int i = 0; i < 8; ++i) { v[i] = __expf(v[i] - mx); sum += v[i]; }
  red[tid] = sum;
  __syncthreads();
  for (int s = 128; s > 0; s >>= 1) {
    if (tid < s) red[tid] += red[tid + s];
    __syncthreads();
  }
  const float inv = 1.0f / red[0];
#pragma unroll
  for (int i = 0; i < 8; ++i) {
    const float pv = v[i] * inv;
    p[tid + i * 256] = pv;
    q[tid + i * 256] = f2bf(pv);
  }
}

// ---------------------------------------------------------------------------
// Kernel 5: E[bn] = P[bn] @ X[b]  (2048x2048 @ 2048x1024 -> f32 out)
// A = probs row-major direct; B = XT (x transposed, [d][t]) direct.
// ---------------------------------------------------------------------------
__global__ __launch_bounds__(256) void gemm_agg(const __bf16* __restrict__ pbf,
                                                const __bf16* __restrict__ xt,
                                                float* __restrict__ e_out) {
  const int bn = blockIdx.z;
  const int b = bn >> 1;
  const int m0 = blockIdx.x * 128;
  const int n0 = blockIdx.y * 64;
  const __bf16* P = pbf + (size_t)bn * T_DIM * T_DIM;
  const __bf16* XT = xt + (size_t)b * D_DIM * T_DIM;  // [d][t]
  float* E = e_out + (size_t)bn * T_DIM * D_DIM;

  const int lane = threadIdx.x & 31;
  const int wave = threadIdx.x >> 5;
  const int wr = wave >> 1, wc = wave & 1;
  const int lm = lane & 15, hi = lane >> 4;

  v8f acc[2][2];
#pragma unroll
  for (int i = 0; i < 2; ++i)
#pragma unroll
    for (int j = 0; j < 2; ++j) acc[i][j] = v8f{0.f, 0.f, 0.f, 0.f, 0.f, 0.f, 0.f, 0.f};

  for (int k0 = 0; k0 < T_DIM; k0 += 32) {
    v16bf afr[2], bfr[2];
#pragma unroll
    for (int mi = 0; mi < 2; ++mi) {
      const int row = m0 + wr * 32 + mi * 16 + lm;
      afr[mi] = load_a_frag(&P[(size_t)row * T_DIM + k0], hi);
    }
#pragma unroll
    for (int ni = 0; ni < 2; ++ni) {
      const int d = n0 + wc * 32 + ni * 16 + lm;
      bfr[ni] = load_b_frag_contig(&XT[(size_t)d * T_DIM + k0 + hi * 16]);
    }
#pragma unroll
    for (int mi = 0; mi < 2; ++mi)
#pragma unroll
      for (int ni = 0; ni < 2; ++ni)
        acc[mi][ni] = __builtin_amdgcn_wmma_f32_16x16x32_bf16(
            false, afr[mi], false, bfr[ni], (short)0, acc[mi][ni], false, false);
  }

#pragma unroll
  for (int mi = 0; mi < 2; ++mi)
#pragma unroll
    for (int ni = 0; ni < 2; ++ni)
#pragma unroll
      for (int r = 0; r < 8; ++r) {
        const int row = m0 + wr * 32 + mi * 16 + (hi ? r + 8 : r);
        const int col = n0 + wc * 32 + ni * 16 + lm;
        E[(size_t)row * D_DIM + col] = acc[mi][ni][r];
      }
}

// ---------------------------------------------------------------------------
// Launch
// ---------------------------------------------------------------------------
extern "C" void kernel_launch(void* const* d_in, const int* in_sizes, int n_in,
                              void* d_out, int out_size, void* d_ws, size_t ws_size,
                              hipStream_t stream) {
  (void)in_sizes; (void)n_in; (void)out_size; (void)ws_size;
  const float* x = (const float*)d_in[0];
  const float* w_qs = (const float*)d_in[1];
  // d_in[2] (w_ks) intentionally unused: reference builds keys from wq too.

  float* e_out = (float*)d_out;                           // [BN, T, D]
  float* attn_out = e_out + (size_t)BNK * T_DIM * D_DIM;  // [BN, T, T]

  const size_t szX  = (size_t)2 * T_DIM * D_DIM;     // 4,194,304 elems
  const size_t szW  = (size_t)2 * D_DIM * IS_DIM;    // 1,048,576 elems
  const size_t szWQ = (size_t)BNK * T_DIM * IS_DIM;  // 4,194,304 elems

  char* ws = (char*)d_ws;
  __bf16* xbf  = (__bf16*)ws;                                   // [b][t][d]
  __bf16* xt   = (__bf16*)(ws + szX * 2);                       // [b][d][t]
  __bf16* wt   = (__bf16*)(ws + szX * 4);                       // [n][h][d]
  __bf16* wqbf = (__bf16*)(ws + szX * 4 + szW * 2);             // [bn][t][h]
  __bf16* pbf  = (__bf16*)(ws + szX * 4 + szW * 2 + szWQ * 2);  // [bn][s][t]

  cvt_bf16<<<(int)((szX + 255) / 256), 256, 0, stream>>>(x, xbf, (int)szX);
  transpose_cvt<<<dim3(D_DIM / 32, T_DIM / 32, 2), 256, 0, stream>>>(x, xt, T_DIM, D_DIM);
  transpose_cvt<<<dim3(IS_DIM / 32, D_DIM / 32, 2), 256, 0, stream>>>(w_qs, wt, D_DIM, IS_DIM);

  gemm_wq<<<dim3(T_DIM / 128, IS_DIM / 64, BNK), 256, 0, stream>>>(xbf, wt, wqbf);
  gemm_attn_logits<<<dim3(T_DIM / 128, T_DIM / 64, BNK), 256, 0, stream>>>(wqbf, attn_out);
  softmax_rows<<<BNK * T_DIM, 256, 0, stream>>>(attn_out, pbf);
  gemm_agg<<<dim3(T_DIM / 128, D_DIM / 64, BNK), 256, 0, stream>>>(pbf, xt, e_out);
}
